// Encoder_31997506355548
// MI455X (gfx1250) — matrix operand order repaired and checked by reading, not compiled, not run on hardware
//
#include <hip/hip_runtime.h>
#include <hip/hip_bf16.h>
#include <stdint.h>

// ---------------------------------------------------------------------------
// Problem constants (from the reference)
// ---------------------------------------------------------------------------
#define VV    32000
#define EE    512
#define HH    1024
#define MIDD  512
#define SS    512
#define BB    128
#define XEE   64
#define FEATT 576   // EE + XEE

typedef __bf16 bf16;
typedef __attribute__((ext_vector_type(16))) __bf16 v16bf;
typedef __attribute__((ext_vector_type(8)))  float  v8f;

union FragU { v16bf v; uint4 q[2]; };

// ---------------------------------------------------------------------------
// WMMA fragment loads (CDNA5 16-bit layouts, CDNA5 ISA §7.12.2)
// A (16x32, MxK): lanes 0-15 = rows, hold K {0..7,16..23}; lanes 16-31 hold
//                 K {8..15,24..31}.  => two contiguous 16B loads per lane.
// B (32x16, KxN): lane n holds column n; lanes 0-15 K {0..7,16..23}; lanes 16-31
//                 K {8..15,24..31}.  => two contiguous 16B loads per lane.
// C = X @ W^T, so B-column n == W row n (W stored [N][K] row-major).
// ---------------------------------------------------------------------------
__device__ __forceinline__ v16bf load_a_frag(const bf16* __restrict__ A, int lda,
                                             int m0, int k0, int lane) {
  const int row  = m0 + (lane & 15);
  const int half = lane >> 4;
  const bf16* base = A + (size_t)row * lda + k0 + half * 8;
  FragU u;
  u.q[0] = *(const uint4*)(base);        // K = k0 + half*8      .. +7
  u.q[1] = *(const uint4*)(base + 16);   // K = k0 + 16 + half*8 .. +7
  return u.v;
}

__device__ __forceinline__ v16bf load_b_frag(const bf16* __restrict__ W, int ldw,
                                             int n0, int k0, int lane) {
  const int col  = n0 + (lane & 15);
  const int half = lane >> 4;
  const bf16* base = W + (size_t)col * ldw + k0 + half * 8;
  FragU u;
  u.q[0] = *(const uint4*)(base);        // K = k0 + half*8      .. +7
  u.q[1] = *(const uint4*)(base + 16);   // K = k0 + 16 + half*8 .. +7
  // Prefetch the weight stream a few K-chunks ahead (global_prefetch_b8).
  __builtin_prefetch((const void*)(base + 128), 0, 1);
  return u.v;
}

// ---------------------------------------------------------------------------
// C[B,N] = A1[B,K1] @ W1[N,K1]^T + A2[B,K2] @ W2[N,K2]^T + bias[N]   (f32 out)
// Wave computes MT M-tiles (MT*16 rows) x 16 cols: each weight (B) fragment is
// loaded ONCE and reused for MT independent WMMAs (8 independent accumulation
// chains -> good XDL ILP, and MT-fold cut in L2 weight traffic).
// Block = 128 threads = 4 waves; grid = (N/64, B/(MT*16)).
// ---------------------------------------------------------------------------
template <int MT>
__global__ void dual_gemm_bias_kernel(const bf16* __restrict__ A1, const bf16* __restrict__ W1, int K1,
                                      const bf16* __restrict__ A2, const bf16* __restrict__ W2, int K2,
                                      const float* __restrict__ bias,
                                      float* __restrict__ C, int N) {
  const int lane = threadIdx.x & 31;
  const int wave = threadIdx.x >> 5;
  const int n0  = blockIdx.x * 64 + wave * 16;
  const int m00 = blockIdx.y * (MT * 16);

  v8f acc[MT];
#pragma unroll
  for (int m = 0; m < MT; ++m) acc[m] = (v8f){0.f, 0.f, 0.f, 0.f, 0.f, 0.f, 0.f, 0.f};

  for (int k = 0; k < K1; k += 32) {
    const v16bf b = load_b_frag(W1, K1, n0, k, lane);
#pragma unroll
    for (int m = 0; m < MT; ++m) {
      const v16bf a = load_a_frag(A1, K1, m00 + m * 16, k, lane);
      acc[m] = __builtin_amdgcn_wmma_f32_16x16x32_bf16(false, a, false, b, (short)0, acc[m], false, false);
    }
  }
  for (int k = 0; k < K2; k += 32) {
    const v16bf b = load_b_frag(W2, K2, n0, k, lane);
#pragma unroll
    for (int m = 0; m < MT; ++m) {
      const v16bf a = load_a_frag(A2, K2, m00 + m * 16, k, lane);
      acc[m] = __builtin_amdgcn_wmma_f32_16x16x32_bf16(false, a, false, b, (short)0, acc[m], false, false);
    }
  }

  const int n  = n0 + (lane & 15);
  const float bv = bias ? bias[n] : 0.0f;
#pragma unroll
  for (int m = 0; m < MT; ++m) {
    const int mb = m00 + m * 16 + (lane >> 4) * 8;
#pragma unroll
    for (int r = 0; r < 8; ++r)
      C[(size_t)(mb + r) * N + n] = acc[m][r] + bv;
  }
}

// ---------------------------------------------------------------------------
// Out[B,N] = bf16( (A1 @ W1^T) * (A2 @ W2^T) )   (elementwise product, m-gate)
// Two accumulator sets are filled in SEQUENTIAL K passes, so MT=8 keeps peak
// live pressure at ~128 acc VGPRs -> full-M blocking, weights streamed once.
// ---------------------------------------------------------------------------
template <int MT>
__global__ void mul_gemm_bf16_kernel(const bf16* __restrict__ A1, const bf16* __restrict__ W1, int K1,
                                     const bf16* __restrict__ A2, const bf16* __restrict__ W2, int K2,
                                     bf16* __restrict__ Out, int N) {
  const int lane = threadIdx.x & 31;
  const int wave = threadIdx.x >> 5;
  const int n0  = blockIdx.x * 64 + wave * 16;
  const int m00 = blockIdx.y * (MT * 16);

  v8f acc1[MT], acc2[MT];
#pragma unroll
  for (int m = 0; m < MT; ++m) {
    acc1[m] = (v8f){0.f, 0.f, 0.f, 0.f, 0.f, 0.f, 0.f, 0.f};
    acc2[m] = (v8f){0.f, 0.f, 0.f, 0.f, 0.f, 0.f, 0.f, 0.f};
  }

  for (int k = 0; k < K1; k += 32) {
    const v16bf b = load_b_frag(W1, K1, n0, k, lane);
#pragma unroll
    for (int m = 0; m < MT; ++m) {
      const v16bf a = load_a_frag(A1, K1, m00 + m * 16, k, lane);
      acc1[m] = __builtin_amdgcn_wmma_f32_16x16x32_bf16(false, a, false, b, (short)0, acc1[m], false, false);
    }
  }
  for (int k = 0; k < K2; k += 32) {
    const v16bf b = load_b_frag(W2, K2, n0, k, lane);
#pragma unroll
    for (int m = 0; m < MT; ++m) {
      const v16bf a = load_a_frag(A2, K2, m00 + m * 16, k, lane);
      acc2[m] = __builtin_amdgcn_wmma_f32_16x16x32_bf16(false, a, false, b, (short)0, acc2[m], false, false);
    }
  }

  const int n = n0 + (lane & 15);
#pragma unroll
  for (int m = 0; m < MT; ++m) {
    const int mb = m00 + m * 16 + (lane >> 4) * 8;
#pragma unroll
    for (int r = 0; r < 8; ++r)
      Out[(size_t)(mb + r) * N + n] = (bf16)(acc1[m][r] * acc2[m][r]);
  }
}

// ---------------------------------------------------------------------------
// Boundary detector.  s[b] = (sigmoid(zb[b]·vs) > 0.5) == (zb[b]·vs > 0).
// One block per batch row, 128 threads, coalesced reads + LDS tree reduction.
// ---------------------------------------------------------------------------
__global__ void boundary_kernel(const float* __restrict__ zb, const float* __restrict__ vs,
                                float* __restrict__ sbuf, float* __restrict__ flags, int t) {
  __shared__ float red[128];
  const int b   = blockIdx.x;
  const int tid = threadIdx.x;
  const float* row = zb + (size_t)b * MIDD;
  float acc = 0.f;
#pragma unroll
  for (int k = tid; k < MIDD; k += 128) acc += row[k] * vs[k];
  red[tid] = acc;
  __syncthreads();
#pragma unroll
  for (int o = 64; o > 0; o >>= 1) {
    if (tid < o) red[tid] += red[tid + o];
    __syncthreads();
  }
  if (tid == 0) {
    const float sv = (red[0] > 0.f) ? 1.0f : 0.0f;   // straight-through binarize
    sbuf[b] = sv;
    if (b == 0) flags[t] = sv;
  }
}

__device__ __forceinline__ float sigmoidf(float x) { return 1.0f / (1.0f + __expf(-x)); }

// cell-1 gates: consumes z1[B,4H], updates c1, emits h1_bf = bf16(h1n*(1-s)),
// x2_bf = bf16(h1n*s)   (input to cell 2).
__global__ void gates1_kernel(const float* __restrict__ z, const float* __restrict__ sbuf,
                              float* __restrict__ c1,
                              bf16* __restrict__ h1_bf, bf16* __restrict__ x2_bf) {
  const int idx = blockIdx.x * blockDim.x + threadIdx.x;   // B*H threads
  const int b = idx >> 10;
  const int j = idx & (HH - 1);
  const float* zr = z + (size_t)b * 4 * HH;
  const float ig = sigmoidf(zr[j]);
  const float fg = sigmoidf(zr[HH + j]);
  const float gg = tanhf(zr[2 * HH + j]);
  const float og = sigmoidf(zr[3 * HH + j]);
  const float cn = fg * c1[idx] + ig * gg;
  const float hn = og * tanhf(cn);
  const float s  = sbuf[b];
  c1[idx]    = cn * (1.0f - s);
  h1_bf[idx] = (bf16)(hn * (1.0f - s));
  x2_bf[idx] = (bf16)(hn * s);
}

// cell-2 gates: whole-batch update gated on batch-0 flag; writes outs[b,t,:].
__global__ void gates2_kernel(const float* __restrict__ z, const float* __restrict__ sbuf,
                              float* __restrict__ c2, float* __restrict__ h2,
                              bf16* __restrict__ h2_bf, float* __restrict__ outs, int t) {
  const int idx = blockIdx.x * blockDim.x + threadIdx.x;   // B*H threads
  const int b = idx >> 10;
  const int j = idx & (HH - 1);
  const float* zr = z + (size_t)b * 4 * HH;
  const float ig = sigmoidf(zr[j]);
  const float fg = sigmoidf(zr[HH + j]);
  const float gg = tanhf(zr[2 * HH + j]);
  const float og = sigmoidf(zr[3 * HH + j]);
  const float cn = fg * c2[idx] + ig * gg;
  const float hn = og * tanhf(cn);
  const bool upd = sbuf[0] > 0.5f;
  const float ho = upd ? hn : h2[idx];
  const float co = upd ? cn : c2[idx];
  h2[idx] = ho;
  c2[idx] = co;
  h2_bf[idx] = (bf16)ho;
  outs[(size_t)b * SS * HH + (size_t)t * HH + j] = ho;
}

// Embedding gather + concat -> time-major bf16 X[S][B][FEAT].
// grid = S*B blocks (s = blk>>7, b = blk&127), 128 threads.
__global__ void embed_kernel(const int* __restrict__ ids, const int* __restrict__ xids,
                             const float* __restrict__ wemb, const float* __restrict__ xemb,
                             bf16* __restrict__ X) {
  const int s = blockIdx.x >> 7;
  const int b = blockIdx.x & (BB - 1);
  const int id  = ids[(size_t)b * SS + s];
  const int xid = xids[(size_t)b * SS + s];
  bf16* dst = X + ((size_t)s * BB + b) * FEATT;
  const float* wr = wemb + (size_t)id * EE;
  for (int j = threadIdx.x; j < EE; j += blockDim.x) dst[j] = (bf16)wr[j];
  const float* xr = xemb + (size_t)xid * XEE;
  for (int j = threadIdx.x; j < XEE; j += blockDim.x) dst[EE + j] = (bf16)xr[j];
}

__global__ void cvt_f32_to_bf16_kernel(const float* __restrict__ src, bf16* __restrict__ dst, int n) {
  const int i = blockIdx.x * blockDim.x + threadIdx.x;
  if (i < n) dst[i] = (bf16)src[i];
}

__global__ void zero_kernel(uint32_t* __restrict__ p, int ndwords) {
  const int i = blockIdx.x * blockDim.x + threadIdx.x;
  if (i < ndwords) p[i] = 0u;
}

// d_out tail: [h2f (B*H) | c2f (B*H)] after final step.
__global__ void finalize_kernel(const float* __restrict__ h2, const float* __restrict__ c2,
                                float* __restrict__ tail) {
  const int i = blockIdx.x * blockDim.x + threadIdx.x;
  if (i < BB * HH)            tail[i] = h2[i];
  else if (i < 2 * BB * HH)   tail[i] = c2[i - BB * HH];
}

// ---------------------------------------------------------------------------
extern "C" void kernel_launch(void* const* d_in, const int* in_sizes, int n_in,
                              void* d_out, int out_size, void* d_ws, size_t ws_size,
                              hipStream_t stream) {
  (void)in_sizes; (void)n_in; (void)out_size; (void)ws_size;

  const int*   enc_in  = (const int*)d_in[0];
  const int*   enc_x   = (const int*)d_in[1];
  const float* wemb    = (const float*)d_in[2];
  const float* xemb    = (const float*)d_in[3];
  const float* Wsi     = (const float*)d_in[4];
  const float* Wsh     = (const float*)d_in[5];
  const float* b_bd    = (const float*)d_in[6];
  const float* vs      = (const float*)d_in[7];
  const float* Wmx1    = (const float*)d_in[8];
  const float* Wmh1    = (const float*)d_in[9];
  const float* Wih1    = (const float*)d_in[10];
  const float* Whh1    = (const float*)d_in[11];
  const float* b1      = (const float*)d_in[12];
  const float* Wmx2    = (const float*)d_in[13];
  const float* Wmh2    = (const float*)d_in[14];
  const float* Wih2    = (const float*)d_in[15];
  const float* Whh2    = (const float*)d_in[16];
  const float* b2      = (const float*)d_in[17];

  // ---- workspace carve-out (all chunks 256B aligned) ----
  char* ws = (char*)d_ws;
  size_t off = 0;
  auto take = [&](size_t bytes) -> char* {
    char* p = ws + off;
    off += (bytes + 255) & ~(size_t)255;
    return p;
  };

  bf16* Xbf    = (bf16*)take((size_t)SS * BB * FEATT * 2);
  bf16* Wsi_b  = (bf16*)take((size_t)MIDD * FEATT * 2);
  bf16* Wsh_b  = (bf16*)take((size_t)MIDD * HH * 2);
  bf16* Wmx1_b = (bf16*)take((size_t)HH * FEATT * 2);
  bf16* Wmh1_b = (bf16*)take((size_t)HH * HH * 2);
  bf16* Wih1_b = (bf16*)take((size_t)4 * HH * FEATT * 2);
  bf16* Whh1_b = (bf16*)take((size_t)4 * HH * HH * 2);
  bf16* Wmx2_b = (bf16*)take((size_t)HH * HH * 2);
  bf16* Wmh2_b = (bf16*)take((size_t)HH * HH * 2);
  bf16* Wih2_b = (bf16*)take((size_t)4 * HH * HH * 2);
  bf16* Whh2_b = (bf16*)take((size_t)4 * HH * HH * 2);

  char*  state0 = ws + off;                       // zeroed as one block
  float* c1    = (float*)take((size_t)BB * HH * 4);
  float* h2    = (float*)take((size_t)BB * HH * 4);
  float* c2    = (float*)take((size_t)BB * HH * 4);
  bf16*  h1_b  = (bf16*)take((size_t)BB * HH * 2);
  bf16*  h2_b  = (bf16*)take((size_t)BB * HH * 2);
  const size_t state_bytes = (size_t)((ws + off) - state0);

  float* zbuf  = (float*)take((size_t)BB * MIDD * 4);
  float* sbuf  = (float*)take((size_t)BB * 4);
  bf16*  m1_b  = (bf16*)take((size_t)BB * HH * 2);
  float* z1    = (float*)take((size_t)BB * 4 * HH * 4);
  bf16*  x2_b  = (bf16*)take((size_t)BB * HH * 2);
  bf16*  m2_b  = (bf16*)take((size_t)BB * HH * 2);
  float* z2    = (float*)take((size_t)BB * 4 * HH * 4);

  float* outs  = (float*)d_out;                                   // [B,S,H]
  float* tail  = outs + (size_t)BB * SS * HH;                     // h2f | c2f
  float* flags = tail + 2 * (size_t)BB * HH;                      // [S]

  // ---- setup: weight conversion, embedding gather, state zero ----
  auto cvt = [&](const float* s, bf16* d, size_t n) {
    cvt_f32_to_bf16_kernel<<<dim3((unsigned)((n + 255) / 256)), 256, 0, stream>>>(s, d, (int)n);
  };
  cvt(Wsi,  Wsi_b,  (size_t)MIDD * FEATT);
  cvt(Wsh,  Wsh_b,  (size_t)MIDD * HH);
  cvt(Wmx1, Wmx1_b, (size_t)HH * FEATT);
  cvt(Wmh1, Wmh1_b, (size_t)HH * HH);
  cvt(Wih1, Wih1_b, (size_t)4 * HH * FEATT);
  cvt(Whh1, Whh1_b, (size_t)4 * HH * HH);
  cvt(Wmx2, Wmx2_b, (size_t)HH * HH);
  cvt(Wmh2, Wmh2_b, (size_t)HH * HH);
  cvt(Wih2, Wih2_b, (size_t)4 * HH * HH);
  cvt(Whh2, Whh2_b, (size_t)4 * HH * HH);

  embed_kernel<<<dim3(SS * BB), 128, 0, stream>>>(enc_in, enc_x, wemb, xemb, Xbf);

  {
    const int ndw = (int)(state_bytes / 4);
    zero_kernel<<<dim3((unsigned)((ndw + 255) / 256)), 256, 0, stream>>>((uint32_t*)state0, ndw);
  }

  // ---- sequential scan over time ----
  const dim3 blk(128);
  const dim3 g_zb(MIDD / 64, 1);             // dual gemm, MT=8 (full M)
  const dim3 g_z(4 * HH / 64, 1);            // dual gemm, MT=8 (full M)
  const dim3 g_m(HH / 64, 1);                // mul gemm,  MT=8 (full M)
  const dim3 g_ew((BB * HH) / 256);          // elementwise: 512 blocks x 256

  for (int t = 0; t < SS; ++t) {
    const bf16* xt = Xbf + (size_t)t * BB * FEATT;

    // zb = x @ Wsi^T + h1 @ Wsh^T + b_bd
    dual_gemm_bias_kernel<8><<<g_zb, blk, 0, stream>>>(xt, Wsi_b, FEATT, h1_b, Wsh_b, HH,
                                                       b_bd, zbuf, MIDD);
    // s[b] = binarize(sigmoid(zb . vs)); flags[t] = s[0]
    boundary_kernel<<<dim3(BB), 128, 0, stream>>>(zbuf, vs, sbuf, flags, t);

    // m1 = (x @ Wmx1^T) * (h1 @ Wmh1^T)
    mul_gemm_bf16_kernel<8><<<g_m, blk, 0, stream>>>(xt, Wmx1_b, FEATT, h1_b, Wmh1_b, HH,
                                                     m1_b, HH);
    // z1 = x @ Wih1^T + m1 @ Whh1^T + b1
    dual_gemm_bias_kernel<8><<<g_z, blk, 0, stream>>>(xt, Wih1_b, FEATT, m1_b, Whh1_b, HH,
                                                      b1, z1, 4 * HH);
    // cell-1 gates; h1 <- h1n*(1-s); x2 = h1n*s
    gates1_kernel<<<g_ew, 256, 0, stream>>>(z1, sbuf, c1, h1_b, x2_b);

    // m2 = (x2 @ Wmx2^T) * (h2 @ Wmh2^T)
    mul_gemm_bf16_kernel<8><<<g_m, blk, 0, stream>>>(x2_b, Wmx2_b, HH, h2_b, Wmh2_b, HH,
                                                     m2_b, HH);
    // z2 = x2 @ Wih2^T + m2 @ Whh2^T + b2
    dual_gemm_bias_kernel<8><<<g_z, blk, 0, stream>>>(x2_b, Wih2_b, HH, m2_b, Whh2_b, HH,
                                                      b2, z2, 4 * HH);
    // cell-2 gates (batch-wide update gated on flag); outs[:, t, :] = h2
    gates2_kernel<<<g_ew, 256, 0, stream>>>(z2, sbuf, c2, h2, h2_b, outs, t);
  }

  finalize_kernel<<<dim3((2 * BB * HH) / 256), 256, 0, stream>>>(h2, c2, tail);
}
